// Act_net_80908593922109
// MI455X (gfx1250) — matrix operand order repaired and checked by reading, not compiled, not run on hardware
//
#include <hip/hip_runtime.h>

typedef __bf16 bf16;
typedef __attribute__((ext_vector_type(16))) __bf16 v16bf;
typedef __attribute__((ext_vector_type(8)))  float   v8f;

constexpr int Bsz = 512, Nn = 128, Ss = 16, Ee = 128, Hh = 512, G4 = 2048;
constexpr float INFV = 1e8f, CLIP = 10.0f;

// ---------------- WMMA fragment loaders (CDNA5 16x16x32 bf16 layouts) ----------------

// A fragment: 16x32 (MxK) bf16, row-major source with leading dim lda.
// lane<16: row=lane,  K = k0+0..7 (vgpr0-3), k0+16..23 (vgpr4-7)
// lane>=16: row=lane-16, K = k0+8..15, k0+24..31
__device__ __forceinline__ v16bf load_a16(const bf16* base, int lda, int k0, int lane) {
  int row = lane & 15;
  int kb  = k0 + ((lane & 16) ? 8 : 0);
  const bf16* p = base + (size_t)row * lda + kb;
  union { v16bf v; uint4 u4[2]; } u;
  u.u4[0] = *reinterpret_cast<const uint4*>(p);
  u.u4[1] = *reinterpret_cast<const uint4*>(p + 16);
  return u.v;
}

// B fragment: 32x16 (KxN) where B[k][n] = W[ntile+n][k], W row-major with leading dim ldw.
// lane<16: col=lane, K=k0+0..7, k0+16..23; lane>=16: col=lane-16, K=k0+8..15, k0+24..31
__device__ __forceinline__ v16bf load_b16(const bf16* Wt, int ldw, int k0, int lane) {
  int col = lane & 15;
  int kb  = k0 + ((lane & 16) ? 8 : 0);
  const bf16* p = Wt + (size_t)col * ldw + kb;
  union { v16bf v; uint4 u4[2]; } u;
  const uint4* q = reinterpret_cast<const uint4*>(p);
  u.u4[0] = q[0];
  u.u4[1] = q[2];
  return u.v;
}

__device__ __forceinline__ v8f wmma_bf16(const v16bf& a, const v16bf& b, const v8f& c) {
  return __builtin_amdgcn_wmma_f32_16x16x32_bf16(false, a, false, b, (short)0, c, false, false);
}

// ---------------- LSTM gate GEMM: gates = A1@W1^T + h@W2^T + bi + bh ----------------
// Register-tiled: one wave computes a 32x64 block (2 M-tiles x 4 N-tiles) of [B, 4H].
// Waves = (B/32)*(4H/64) = 16*32 = 512.
__global__ __launch_bounds__(256) void lstm_gates_kernel(
    const bf16* __restrict__ A1, int lda1,
    const bf16* __restrict__ W1,
    const bf16* __restrict__ A2,
    const bf16* __restrict__ W2,
    const float* __restrict__ bi, const float* __restrict__ bh,
    float* __restrict__ gates) {
  int wave = (blockIdx.x * blockDim.x + threadIdx.x) >> 5;
  int lane = threadIdx.x & 31;
  const int nq = G4 / 64;                    // 32 N-quads
  int mp = wave / nq, q = wave % nq;
  int m0 = mp * 32, n0 = q * 64;
  v8f acc[2][4] = {};
  for (int k0 = 0; k0 < Ee; k0 += 32) {
    v16bf a0 = load_a16(A1 + (size_t)m0 * lda1,        lda1, k0, lane);
    v16bf a1 = load_a16(A1 + (size_t)(m0 + 16) * lda1, lda1, k0, lane);
#pragma unroll
    for (int j = 0; j < 4; ++j) {
      v16bf bfr = load_b16(W1 + (size_t)(n0 + j * 16) * Ee, Ee, k0, lane);
      acc[0][j] = wmma_bf16(a0, bfr, acc[0][j]);
      acc[1][j] = wmma_bf16(a1, bfr, acc[1][j]);
    }
  }
  for (int k0 = 0; k0 < Hh; k0 += 32) {
    v16bf a0 = load_a16(A2 + (size_t)m0 * Hh,        Hh, k0, lane);
    v16bf a1 = load_a16(A2 + (size_t)(m0 + 16) * Hh, Hh, k0, lane);
#pragma unroll
    for (int j = 0; j < 4; ++j) {
      v16bf bfr = load_b16(W2 + (size_t)(n0 + j * 16) * Hh, Hh, k0, lane);
      acc[0][j] = wmma_bf16(a0, bfr, acc[0][j]);
      acc[1][j] = wmma_bf16(a1, bfr, acc[1][j]);
    }
  }
  int col = lane & 15;
  int rb  = (lane & 16) ? 8 : 0;
#pragma unroll
  for (int i = 0; i < 2; ++i)
#pragma unroll
    for (int j = 0; j < 4; ++j) {
      int jc = n0 + j * 16 + col;
      float bias = bi[jc] + bh[jc];
      float* Crow = gates + (size_t)(m0 + i * 16) * G4 + jc;
#pragma unroll
      for (int r = 0; r < 8; ++r)
        Crow[(size_t)(rb + r) * G4] = acc[i][j][r] + bias;
    }
}

// ---------------- [M,512] @ [512,512]^T GEMM, register-tiled 32x64 per wave ----------------
template <typename OutT>
__device__ __forceinline__ void gemm_h512_2x4(const bf16* __restrict__ A,
                                              const bf16* __restrict__ W,
                                              const float* __restrict__ bias,
                                              OutT* __restrict__ Out,
                                              int wave, int lane) {
  const int nq = Hh / 64;                    // 8 N-quads
  int mp = wave / nq, q = wave % nq;
  int m0 = mp * 32, n0 = q * 64;
  v8f acc[2][4] = {};
  for (int k0 = 0; k0 < Hh; k0 += 32) {
    v16bf a0 = load_a16(A + (size_t)m0 * Hh,        Hh, k0, lane);
    v16bf a1 = load_a16(A + (size_t)(m0 + 16) * Hh, Hh, k0, lane);
#pragma unroll
    for (int j = 0; j < 4; ++j) {
      v16bf bfr = load_b16(W + (size_t)(n0 + j * 16) * Hh, Hh, k0, lane);
      acc[0][j] = wmma_bf16(a0, bfr, acc[0][j]);
      acc[1][j] = wmma_bf16(a1, bfr, acc[1][j]);
    }
  }
  int col = lane & 15;
  int rb  = (lane & 16) ? 8 : 0;
#pragma unroll
  for (int i = 0; i < 2; ++i)
#pragma unroll
    for (int j = 0; j < 4; ++j) {
      float bb = bias[n0 + j * 16 + col];
      OutT* Crow = Out + (size_t)(m0 + i * 16) * Hh + n0 + j * 16 + col;
#pragma unroll
      for (int r = 0; r < 8; ++r)
        Crow[(size_t)(rb + r) * Hh] = (OutT)(acc[i][j][r] + bb);
    }
}

// Out bf16: u2g/u2p projections (keeps decoder attention working set L2-resident).
__global__ __launch_bounds__(256) void gemm_h512_bf16out_kernel(
    const bf16* __restrict__ A, const bf16* __restrict__ W,
    const float* __restrict__ bias, bf16* __restrict__ Out) {
  gemm_h512_2x4<bf16>(A, W, bias, Out,
                      (blockIdx.x * blockDim.x + threadIdx.x) >> 5, threadIdx.x & 31);
}

// Out f32: q@Wq^T + bq and q@Wq2^T + bq2.
__global__ __launch_bounds__(256) void gemm_h512_f32out_kernel(
    const bf16* __restrict__ A, const bf16* __restrict__ W,
    const float* __restrict__ bias, float* __restrict__ Out) {
  gemm_h512_2x4<float>(A, W, bias, Out,
                       (blockIdx.x * blockDim.x + threadIdx.x) >> 5, threadIdx.x & 31);
}

// ---------------- Embedding: emb_bf[b,n,:] = x[b,n,:] @ W_emb ----------------
__global__ void emb_kernel(const float* __restrict__ x, const float* __restrict__ W_emb,
                           bf16* __restrict__ emb_bf) {
  int bn = blockIdx.x;            // B*N blocks
  int e  = threadIdx.x;           // E = 128 threads
  __shared__ float xs[Ss];
  if (e < Ss) xs[e] = x[(size_t)bn * Ss + e];
  __syncthreads();
  float acc = 0.f;
#pragma unroll
  for (int s = 0; s < Ss; ++s) acc += xs[s] * W_emb[s * Ee + e];
  emb_bf[(size_t)bn * Ee + e] = (bf16)acc;
}

// ---------------- LSTM cell elementwise ----------------
__global__ void lstm_cell_kernel(const float* __restrict__ gates, float* __restrict__ c,
                                 bf16* __restrict__ h_bf, bf16* __restrict__ ref_slice,
                                 long long ref_stride) {
  int idx = blockIdx.x * blockDim.x + threadIdx.x;   // B*H
  int b = idx / Hh, hh = idx - b * Hh;
  const float* g = gates + (size_t)b * G4;
  float ig = 1.f / (1.f + expf(-g[hh]));
  float fg = 1.f / (1.f + expf(-g[Hh + hh]));
  float gg = tanhf(g[2 * Hh + hh]);
  float og = 1.f / (1.f + expf(-g[3 * Hh + hh]));
  float cn = fg * c[idx] + ig * gg;
  c[idx] = cn;
  float hn = og * tanhf(cn);
  h_bf[idx] = (bf16)hn;
  if (ref_slice) ref_slice[(size_t)b * ref_stride + hh] = (bf16)hn;
}

// ---------------- Glimpse attention: one block per batch row ----------------
__global__ void glimpse_kernel(const float* __restrict__ qW, const bf16* __restrict__ u2g,
                               const float* __restrict__ Vec, const float* __restrict__ mask,
                               const bf16* __restrict__ ref, bf16* __restrict__ q_bf) {
  int b = blockIdx.x, tid = threadIdx.x;   // 128 threads, tid = n
  __shared__ float sq[Hh], sv[Hh], red[Nn], sa[Nn];
  // Prefetch the ref row this block will sweep after the softmax barriers.
  __builtin_prefetch(ref + ((size_t)b * Nn + tid) * Hh, 0, 3);
  for (int i = tid; i < Hh; i += Nn) { sq[i] = qW[(size_t)b * Hh + i]; sv[i] = Vec[i]; }
  __syncthreads();
  const bf16* row = u2g + ((size_t)b * Nn + tid) * Hh;
  float u = 0.f;
  for (int h = 0; h < Hh; ++h) u += tanhf(sq[h] + (float)row[h]) * sv[h];
  u -= INFV * mask[(size_t)b * Nn + tid];
  red[tid] = u; __syncthreads();
  for (int s = 64; s > 0; s >>= 1) { if (tid < s) red[tid] = fmaxf(red[tid], red[tid + s]); __syncthreads(); }
  float mx = red[0]; __syncthreads();
  float e = expf(u - mx);
  red[tid] = e; __syncthreads();
  for (int s = 64; s > 0; s >>= 1) { if (tid < s) red[tid] += red[tid + s]; __syncthreads(); }
  float sum = red[0]; __syncthreads();
  sa[tid] = e / sum; __syncthreads();
#pragma unroll
  for (int hc = 0; hc < Hh / Nn; ++hc) {
    int h = tid + hc * Nn;
    float s = 0.f;
    for (int n = 0; n < Nn; ++n) s += sa[n] * (float)ref[((size_t)b * Nn + n) * Hh + h];
    q_bf[(size_t)b * Hh + h] = (bf16)s;
  }
}

// ---------------- Pointer head: logits, log-softmax, ps write, teacher forcing ----------------
__global__ void pointer_kernel(const float* __restrict__ qW2, const bf16* __restrict__ u2p,
                               const float* __restrict__ Vec2, float* __restrict__ mask,
                               const bf16* __restrict__ emb_bf, const int* __restrict__ y,
                               int t, float* __restrict__ ps, bf16* __restrict__ dec_in) {
  int b = blockIdx.x, tid = threadIdx.x;   // 128 threads, tid = n
  __shared__ float sq[Hh], sv[Hh], red[Nn];
  int yt = y[(size_t)b * Nn + t];
  // Prefetch the teacher-forced embedding row gathered after the reductions.
  __builtin_prefetch(emb_bf + ((size_t)b * Nn + yt) * Ee, 0, 3);
  for (int i = tid; i < Hh; i += Nn) { sq[i] = qW2[(size_t)b * Hh + i]; sv[i] = Vec2[i]; }
  __syncthreads();
  const bf16* row = u2p + ((size_t)b * Nn + tid) * Hh;
  float u = 0.f;
  for (int h = 0; h < Hh; ++h) u += tanhf(sq[h] + (float)row[h]) * sv[h];
  u = CLIP * tanhf(u);
  u -= INFV * mask[(size_t)b * Nn + tid];
  red[tid] = u; __syncthreads();
  for (int s = 64; s > 0; s >>= 1) { if (tid < s) red[tid] = fmaxf(red[tid], red[tid + s]); __syncthreads(); }
  float mx = red[0]; __syncthreads();
  float e = expf(u - mx);
  red[tid] = e; __syncthreads();
  for (int s = 64; s > 0; s >>= 1) { if (tid < s) red[tid] += red[tid + s]; __syncthreads(); }
  float sum = red[0]; __syncthreads();
  float lp = (u - mx) - logf(sum);
  ps[((size_t)b * Nn + t) * Nn + tid] = lp;
  dec_in[(size_t)b * Ee + tid] = emb_bf[((size_t)b * Nn + yt) * Ee + tid];  // E == 128 == blockDim
  __syncthreads();
  if (tid == 0) mask[(size_t)b * Nn + yt] += 1.0f;
}

// ---------------- Misc small kernels ----------------
__global__ void cvt_kernel(const float* __restrict__ src, bf16* __restrict__ dst, int n) {
  int i = blockIdx.x * blockDim.x + threadIdx.x;
  if (i < n) dst[i] = (bf16)src[i];
}

__global__ void init_decin_kernel(const float* __restrict__ dec0, bf16* __restrict__ dec_in) {
  int i = blockIdx.x * blockDim.x + threadIdx.x;
  if (i < Bsz * Ee) dec_in[i] = (bf16)dec0[i & (Ee - 1)];
}

__global__ void finalize_kernel(const int* __restrict__ y, const float* __restrict__ ps,
                                float* __restrict__ out_y, float* __restrict__ out_ll) {
  int b = blockIdx.x, tid = threadIdx.x;   // 128 threads, tid = t
  __shared__ float red[Nn];
  int yv = y[(size_t)b * Nn + tid];
  out_y[(size_t)b * Nn + tid] = (float)yv;
  red[tid] = ps[((size_t)b * Nn + tid) * Nn + yv];
  __syncthreads();
  for (int s = 64; s > 0; s >>= 1) { if (tid < s) red[tid] += red[tid + s]; __syncthreads(); }
  if (tid == 0) out_ll[b] = red[0];
}

// ---------------- Host launcher ----------------
extern "C" void kernel_launch(void* const* d_in, const int* in_sizes, int n_in,
                              void* d_out, int out_size, void* d_ws, size_t ws_size,
                              hipStream_t stream) {
  (void)in_sizes; (void)n_in; (void)out_size; (void)ws_size;
  const float* x      = (const float*)d_in[0];
  const int*   y      = (const int*)  d_in[1];
  const float* W_emb  = (const float*)d_in[2];
  const float* Wi_enc = (const float*)d_in[3];
  const float* Wh_enc = (const float*)d_in[4];
  const float* bi_enc = (const float*)d_in[5];
  const float* bh_enc = (const float*)d_in[6];
  const float* Wi_dec = (const float*)d_in[7];
  const float* Wh_dec = (const float*)d_in[8];
  const float* bi_dec = (const float*)d_in[9];
  const float* bh_dec = (const float*)d_in[10];
  const float* Wq     = (const float*)d_in[11];
  const float* bq     = (const float*)d_in[12];
  const float* Wref   = (const float*)d_in[13];
  const float* bref   = (const float*)d_in[14];
  const float* Wq2    = (const float*)d_in[15];
  const float* bq2    = (const float*)d_in[16];
  const float* Wref2  = (const float*)d_in[17];
  const float* bref2  = (const float*)d_in[18];
  const float* Vec    = (const float*)d_in[19];
  const float* Vec2   = (const float*)d_in[20];
  const float* dec0   = (const float*)d_in[21];

  // Workspace carve-out (~233 MB): bf16 activations/weights + f32 state.
  char* w = (char*)d_ws;
  auto take = [&](size_t bytes) { char* p = w; w += (bytes + 255) & ~(size_t)255; return p; };
  bf16* emb_bf    = (bf16*)take((size_t)Bsz * Nn * Ee * 2);
  bf16* ref_bf    = (bf16*)take((size_t)Bsz * Nn * Hh * 2);
  bf16* u2g_bf    = (bf16*)take((size_t)Bsz * Nn * Hh * 2);
  bf16* u2p_bf    = (bf16*)take((size_t)Bsz * Nn * Hh * 2);
  bf16* Wi_enc_bf = (bf16*)take((size_t)G4 * Ee * 2);
  bf16* Wh_enc_bf = (bf16*)take((size_t)G4 * Hh * 2);
  bf16* Wi_dec_bf = (bf16*)take((size_t)G4 * Ee * 2);
  bf16* Wh_dec_bf = (bf16*)take((size_t)G4 * Hh * 2);
  bf16* Wq_bf     = (bf16*)take((size_t)Hh * Hh * 2);
  bf16* Wq2_bf    = (bf16*)take((size_t)Hh * Hh * 2);
  bf16* Wref_bf   = (bf16*)take((size_t)Hh * Hh * 2);
  bf16* Wref2_bf  = (bf16*)take((size_t)Hh * Hh * 2);
  bf16* h_bf      = (bf16*)take((size_t)Bsz * Hh * 2);
  float* c_st     = (float*)take((size_t)Bsz * Hh * 4);
  bf16* q_bf      = (bf16*)take((size_t)Bsz * Hh * 2);
  float* gates    = (float*)take((size_t)Bsz * G4 * 4);
  float* qW       = (float*)take((size_t)Bsz * Hh * 4);
  float* mask     = (float*)take((size_t)Bsz * Nn * 4);
  bf16* dec_in    = (bf16*)take((size_t)Bsz * Ee * 2);

  float* out_y  = (float*)d_out;
  float* out_ll = out_y + (size_t)Bsz * Nn;
  float* ps     = out_ll + Bsz;

  auto cvt = [&](const float* s, bf16* d, int n) {
    cvt_kernel<<<(n + 255) / 256, 256, 0, stream>>>(s, d, n);
  };
  cvt(Wi_enc, Wi_enc_bf, G4 * Ee);
  cvt(Wh_enc, Wh_enc_bf, G4 * Hh);
  cvt(Wi_dec, Wi_dec_bf, G4 * Ee);
  cvt(Wh_dec, Wh_dec_bf, G4 * Hh);
  cvt(Wq,     Wq_bf,     Hh * Hh);
  cvt(Wq2,    Wq2_bf,    Hh * Hh);
  cvt(Wref,   Wref_bf,   Hh * Hh);
  cvt(Wref2,  Wref2_bf,  Hh * Hh);

  hipMemsetAsync(h_bf, 0, (size_t)Bsz * Hh * 2, stream);
  hipMemsetAsync(c_st, 0, (size_t)Bsz * Hh * 4, stream);
  hipMemsetAsync(mask, 0, (size_t)Bsz * Nn * 4, stream);

  emb_kernel<<<Bsz * Nn, Ee, 0, stream>>>(x, W_emb, emb_bf);

  // Encoder: 128 sequential steps; gate GEMM [512 x 640] @ [640 x 2048] via WMMA.
  // 512 waves of 32x64 register tiles -> 64 blocks of 256 threads.
  for (int t = 0; t < Nn; ++t) {
    lstm_gates_kernel<<<64, 256, 0, stream>>>(emb_bf + (size_t)t * Ee, Nn * Ee,
                                              Wi_enc_bf, h_bf, Wh_enc_bf,
                                              bi_enc, bh_enc, gates);
    lstm_cell_kernel<<<(Bsz * Hh) / 256, 256, 0, stream>>>(gates, c_st, h_bf,
                                                           ref_bf + (size_t)t * Hh,
                                                           (long long)Nn * Hh);
  }

  // Loop-invariant attention keys: [65536 x 512] @ [512 x 512] each, bf16 out.
  // (65536/32)*(512/64) = 16384 waves -> 2048 blocks.
  gemm_h512_bf16out_kernel<<<2048, 256, 0, stream>>>(ref_bf, Wref_bf,  bref,  u2g_bf);
  gemm_h512_bf16out_kernel<<<2048, 256, 0, stream>>>(ref_bf, Wref2_bf, bref2, u2p_bf);

  init_decin_kernel<<<(Bsz * Ee) / 256, 256, 0, stream>>>(dec0, dec_in);

  // Decoder: 128 sequential steps.  Query GEMMs: (512/32)*(512/64)=128 waves -> 16 blocks.
  for (int t = 0; t < Nn; ++t) {
    lstm_gates_kernel<<<64, 256, 0, stream>>>(dec_in, Ee,
                                              Wi_dec_bf, h_bf, Wh_dec_bf,
                                              bi_dec, bh_dec, gates);
    lstm_cell_kernel<<<(Bsz * Hh) / 256, 256, 0, stream>>>(gates, c_st, h_bf, nullptr, 0);
    gemm_h512_f32out_kernel<<<16, 256, 0, stream>>>(h_bf, Wq_bf, bq, qW);
    glimpse_kernel<<<Bsz, Nn, 0, stream>>>(qW, u2g_bf, Vec, mask, ref_bf, q_bf);
    gemm_h512_f32out_kernel<<<16, 256, 0, stream>>>(q_bf, Wq2_bf, bq2, qW);
    pointer_kernel<<<Bsz, Nn, 0, stream>>>(qW, u2p_bf, Vec2, mask, emb_bf, y, t, ps, dec_in);
  }

  finalize_kernel<<<Bsz, Nn, 0, stream>>>(y, ps, out_y, out_ll);
}
